// TemporalBlock_19121194401967
// MI455X (gfx1250) — matrix operand order repaired and checked by reading, not compiled, not run on hardware
//
#include <hip/hip_runtime.h>
#include <math.h>

// ---------------------------------------------------------------------------
// TemporalBlock for MI455X (gfx1250): bf16 WMMA GEMMs + WMMA attention.
// GEMMs: double-buffered LDS fed by GLOBAL_LOAD_ASYNC_TO_LDS_B128 (ASYNCcnt),
// compute on v_wmma_f32_16x16x32_bf16 (f32 accumulate).
// ---------------------------------------------------------------------------

#define DIMC   768
#define NHEAD  12
#define HDIM   64
#define MLPD   3072
#define NTOK   16384      // B*H*W = 16*32*32
#define NSPAT  1024       // H*W
#define NFRM   16         // temporal axis (B)

typedef __bf16 bf16;
typedef __attribute__((ext_vector_type(16))) __bf16 bf16x16;
typedef __attribute__((ext_vector_type(8)))  __bf16 bf16x8;
typedef __attribute__((ext_vector_type(8)))  float  f32x8;

static __device__ __forceinline__ bf16 f2bf(float f) {
    union { float f; unsigned u; } v; v.f = f;
    unsigned r = v.u + 0x7FFFu + ((v.u >> 16) & 1u);   // round-to-nearest-even
    union { unsigned short s; bf16 b; } o; o.s = (unsigned short)(r >> 16);
    return o.b;
}

static __device__ __forceinline__ f32x8 zero8() {
    f32x8 v;
#pragma unroll
    for (int i = 0; i < 8; ++i) v[i] = 0.f;
    return v;
}

// Async global -> LDS 16B copy (per-lane addresses), tracked by ASYNCcnt.
// LDS byte address = low 32 bits of the generic pointer (aperture mapping).
static __device__ __forceinline__ void async_b128(void* lds, const void* gaddr) {
    unsigned loff = (unsigned)(size_t)lds;
    unsigned long long ga = (unsigned long long)(size_t)gaddr;
    asm volatile("global_load_async_to_lds_b128 %0, %1, off"
                 :: "v"(loff), "v"(ga) : "memory");
}
static __device__ __forceinline__ void wait_async0() {
    asm volatile("s_wait_asynccnt 0x0" ::: "memory");
}

// A-fragment (16x32 bf16, M x K): lane (r=lane%16, h=lane/16) holds row r,
// K elements [16h .. 16h+15].  rowp points at mat[r][kbase].
static __device__ __forceinline__ bf16x16 frag_a(const bf16* rowp, int lh) {
    const bf16* p = rowp + lh * 16;
    bf16x8 a0 = *(const bf16x8*)p;
    bf16x8 a1 = *(const bf16x8*)(p + 8);
    bf16x16 t;
#pragma unroll
    for (int e = 0; e < 8; ++e) { t[e] = a0[e]; t[e + 8] = a1[e]; }
    return t;
}

// B-fragment (32x16 bf16, K x N): lane (n=lane%16, h=lane/16) holds column n,
// K elements [16h .. 16h+15].  colp points at storage where that column's K
// values are contiguous.
static __device__ __forceinline__ bf16x16 frag_b(const bf16* colp, int lh) {
    const bf16* p = colp + lh * 16;
    bf16x8 b0 = *(const bf16x8*)p;
    bf16x8 b1 = *(const bf16x8*)(p + 8);
    bf16x16 t;
#pragma unroll
    for (int e = 0; e < 8; ++e) { t[e] = b0[e]; t[e + 8] = b1[e]; }
    return t;
}

// ---------------------------------------------------------------------------
// f32 -> bf16 weight conversion
// ---------------------------------------------------------------------------
__global__ void cvt_kernel(const float* __restrict__ in, bf16* __restrict__ out, int n) {
    int i = blockIdx.x * blockDim.x + threadIdx.x;
    if (i < n) out[i] = f2bf(in[i]);
}

// ---------------------------------------------------------------------------
// LayerNorm (768) -> bf16.  One wave per token, 8 tokens per block.
// ---------------------------------------------------------------------------
__global__ __launch_bounds__(256) void ln_kernel(const float* __restrict__ x,
                                                 const float* __restrict__ g,
                                                 const float* __restrict__ bta,
                                                 bf16* __restrict__ out) {
    int wid = threadIdx.x >> 5, lane = threadIdx.x & 31;
    int tok = blockIdx.x * 8 + wid;
    const float* px = x + (size_t)tok * DIMC;

    float4 vv[6];
    float s = 0.f, s2 = 0.f;
#pragma unroll
    for (int it = 0; it < 6; ++it) {
        vv[it] = *(const float4*)(px + (lane + it * 32) * 4);
        s  += vv[it].x + vv[it].y + vv[it].z + vv[it].w;
        s2 += vv[it].x * vv[it].x + vv[it].y * vv[it].y
            + vv[it].z * vv[it].z + vv[it].w * vv[it].w;
    }
#pragma unroll
    for (int off = 16; off > 0; off >>= 1) {
        s  += __shfl_xor(s,  off, 32);
        s2 += __shfl_xor(s2, off, 32);
    }
    float mu  = s * (1.f / DIMC);
    float var = s2 * (1.f / DIMC) - mu * mu;
    float rs  = rsqrtf(var + 1e-5f);

    bf16* po = out + (size_t)tok * DIMC;
#pragma unroll
    for (int it = 0; it < 6; ++it) {
        int idx = (lane + it * 32) * 4;
        float4 w4 = *(const float4*)(g + idx);
        float4 b4 = *(const float4*)(bta + idx);
        union { bf16 h[4]; uint2 u; } pk;
        pk.h[0] = f2bf((vv[it].x - mu) * rs * w4.x + b4.x);
        pk.h[1] = f2bf((vv[it].y - mu) * rs * w4.y + b4.y);
        pk.h[2] = f2bf((vv[it].z - mu) * rs * w4.z + b4.z);
        pk.h[3] = f2bf((vv[it].w - mu) * rs * w4.w + b4.w);
        *(uint2*)(po + idx) = pk.u;
    }
}

// ---------------------------------------------------------------------------
// GEMM: C[row, col] = sum_k A[row,k] * W[col,k] + bias[col]  (A: NTOKxK bf16,
// W: McolsxK bf16 row-major).  128x128x32 tiles, double-buffered LDS fed by
// async global->LDS; 8 waves, wave = 64x32 patch = 4x2 WMMA accumulators.
// MODE 0: -> bf16       MODE 1: -> f32
// MODE 2: GELU -> bf16  MODE 3: + resid -> f32
// ---------------------------------------------------------------------------
template<int MODE>
__global__ __launch_bounds__(256) void gemm_bf16(const bf16*  __restrict__ A,
                                                 const bf16*  __restrict__ Wm,
                                                 const float* __restrict__ bias,
                                                 const float* __restrict__ resid,
                                                 void* __restrict__ Cout,
                                                 int Mcols, int K) {
    __shared__ alignas(16) bf16 As [2][128 * 40];
    __shared__ alignas(16) bf16 Wsh[2][128 * 40];

    int tid = threadIdx.x;
    int lane = tid & 31, wid = tid >> 5;
    int waveM = wid & 1, waveN = wid >> 1;          // 2 x 4 wave grid
    int rowBlk = blockIdx.y * 128, colBlk = blockIdx.x * 128;
    int lrow = tid >> 1, lhalf = tid & 1;           // 2 threads/row, 32B each
    int lc = lane & 15, lh = lane >> 4;

    f32x8 acc[4][2];
#pragma unroll
    for (int i = 0; i < 4; ++i)
#pragma unroll
        for (int j = 0; j < 2; ++j) acc[i][j] = zero8();

    const bf16* pAg = A  + (size_t)(rowBlk + lrow) * K + lhalf * 16;
    const bf16* pWg = Wm + (size_t)(colBlk + lrow) * K + lhalf * 16;
    int ldsIdx = lrow * 40 + lhalf * 16;            // this thread's LDS slot
    int nk = K >> 5;

    // prologue: async-fill buffer 0 with K-chunk 0
    async_b128(&As [0][ldsIdx],     pAg);
    async_b128(&As [0][ldsIdx + 8], pAg + 8);
    async_b128(&Wsh[0][ldsIdx],     pWg);
    async_b128(&Wsh[0][ldsIdx + 8], pWg + 8);

    for (int kc = 0; kc < nk; ++kc) {
        int p = kc & 1;
        wait_async0();          // my loads into buf p are done
        __syncthreads();        // everyone's are; everyone done reading buf 1-p

        if (kc + 1 < nk) {      // async-fill the other buffer while computing
            const bf16* ga = pAg + (kc + 1) * 32;
            const bf16* gw = pWg + (kc + 1) * 32;
            async_b128(&As [1 - p][ldsIdx],     ga);
            async_b128(&As [1 - p][ldsIdx + 8], ga + 8);
            async_b128(&Wsh[1 - p][ldsIdx],     gw);
            async_b128(&Wsh[1 - p][ldsIdx + 8], gw + 8);
        }

        bf16x16 bfrag[2];
#pragma unroll
        for (int nt = 0; nt < 2; ++nt)
            bfrag[nt] = frag_b(&Wsh[p][(waveN * 32 + nt * 16 + lc) * 40], lh);
        bf16x16 afrag[4];
#pragma unroll
        for (int mt = 0; mt < 4; ++mt)
            afrag[mt] = frag_a(&As[p][(waveM * 64 + mt * 16 + lc) * 40], lh);

#pragma unroll
        for (int mt = 0; mt < 4; ++mt)
#pragma unroll
            for (int nt = 0; nt < 2; ++nt)
                acc[mt][nt] = __builtin_amdgcn_wmma_f32_16x16x32_bf16(
                    false, afrag[mt], false, bfrag[nt], (short)0, acc[mt][nt],
                    false, false);
    }

    // Epilogue.  D lane (vi, lh) -> (M = vi + 8*lh, N = lane%16)
#pragma unroll
    for (int nt = 0; nt < 2; ++nt) {
        int col = colBlk + waveN * 32 + nt * 16 + lc;
        float bv = bias[col];
#pragma unroll
        for (int mt = 0; mt < 4; ++mt) {
#pragma unroll
            for (int vi = 0; vi < 8; ++vi) {
                int row = rowBlk + waveM * 64 + mt * 16 + vi + lh * 8;
                float v = acc[mt][nt][vi] + bv;
                size_t off = (size_t)row * Mcols + col;
                if (MODE == 0) {
                    ((bf16*)Cout)[off] = f2bf(v);
                } else if (MODE == 1) {
                    ((float*)Cout)[off] = v;
                } else if (MODE == 2) {
                    v = 0.5f * v * (1.f + erff(v * 0.70710678118f));
                    ((bf16*)Cout)[off] = f2bf(v);
                } else {
                    ((float*)Cout)[off] = v + resid[off];
                }
            }
        }
    }
}

// ---------------------------------------------------------------------------
// Temporal attention: one wave per (spatial pos, head) sequence (16 frames).
// scores = scale*q.k^T (2 WMMA) ; bias P = q @ rel_pos^T (4 WMMA, padded 31->32)
// softmax scalar ; out = attn @ v (4 WMMA, attn zero-padded K 16->32).
// ---------------------------------------------------------------------------
struct alignas(16) AttnSmem {
    bf16  q[16 * 64];      // [frame][d]
    bf16  k[16 * 64];      // [frame][d]
    bf16  vT[64 * 32];     // [d][frame], frames 16..31 zero
    bf16  attn[16 * 32];   // [q][k], k 16..31 zero
    float sc[16 * 16];
    float P[16 * 32];
};

__global__ __launch_bounds__(128) void attn_kernel(const bf16*  __restrict__ qkv,
                                                   const float* __restrict__ rel_pos,
                                                   bf16* __restrict__ out) {
    __shared__ alignas(16) bf16 relp[32 * 64];   // rel_pos bf16, row 31 zero
    __shared__ AttnSmem sm[4];

    int tid = threadIdx.x, lane = tid & 31, wid = tid >> 5;
    int lc = lane & 15, lh = lane >> 4;

    // rel_pos (31x64 f32) -> bf16 LDS, shared by all 4 waves; pad row 31 = 0
    for (int it = 0; it < 16; ++it) {
        int e = tid + it * 128;
        int r = e >> 6, d = e & 63;
        relp[e] = (r < 31) ? f2bf(rel_pos[r * 64 + d]) : f2bf(0.f);
    }
    __syncthreads();

    AttnSmem& wm = sm[wid];
    int seq  = blockIdx.x * 4 + wid;
    int head = seq >> 10;          // 12288 = 12 * 1024
    int s    = seq & 1023;
    const bf16* base = qkv + (size_t)s * (3 * DIMC) + head * HDIM;

    // zero vT frame-pad columns 16..31
    for (int it = 0; it < 32; ++it) {
        int e = lane + it * 32;                     // 0..1023
        wm.vT[(e >> 4) * 32 + 16 + (e & 15)] = f2bf(0.f);
    }
    // load q, k row-major; v transposed  (uint2 = 4 bf16 per load)
    union U2 { uint2 u; bf16 h[4]; };
    for (int it = 0; it < 8; ++it) {
        int e4 = (lane + it * 32) * 4;              // 0..1020
        int f = e4 >> 6, d = e4 & 63;
        const bf16* src = base + (size_t)f * NSPAT * (3 * DIMC) + d;
        *(uint2*)&wm.q[f * 64 + d] = *(const uint2*)(src);
        *(uint2*)&wm.k[f * 64 + d] = *(const uint2*)(src + DIMC);
        U2 vv; vv.u = *(const uint2*)(src + 2 * DIMC);
        wm.vT[(d + 0) * 32 + f] = vv.h[0];
        wm.vT[(d + 1) * 32 + f] = vv.h[1];
        wm.vT[(d + 2) * 32 + f] = vv.h[2];
        wm.vT[(d + 3) * 32 + f] = vv.h[3];
    }

    // scores (q.k^T) and rel-pos projection P = q @ rel_pos^T, K = 64 = 2x32
    f32x8 accS  = zero8();
    f32x8 accP0 = zero8();
    f32x8 accP1 = zero8();
#pragma unroll
    for (int c = 0; c < 2; ++c) {
        bf16x16 aq = frag_a(&wm.q[lc * 64 + c * 32], lh);
        accS  = __builtin_amdgcn_wmma_f32_16x16x32_bf16(
            false, aq, false, frag_b(&wm.k[lc * 64 + c * 32], lh),
            (short)0, accS, false, false);
        accP0 = __builtin_amdgcn_wmma_f32_16x16x32_bf16(
            false, aq, false, frag_b(&relp[lc * 64 + c * 32], lh),
            (short)0, accP0, false, false);
        accP1 = __builtin_amdgcn_wmma_f32_16x16x32_bf16(
            false, aq, false, frag_b(&relp[(16 + lc) * 64 + c * 32], lh),
            (short)0, accP1, false, false);
    }
    const float scale = 0.125f;   // HDIM^-0.5
#pragma unroll
    for (int vi = 0; vi < 8; ++vi) {
        int i = vi + lh * 8;
        wm.sc[i * 16 + lc]      = accS[vi] * scale;
        wm.P [i * 32 + lc]      = accP0[vi];
        wm.P [i * 32 + 16 + lc] = accP1[vi];
    }

    // scalar softmax, bias gathered as P[i][i-j+15]; lanes 0..15 = rows
    if (lane < 16) {
        int i = lane;
        float sv[16];
        float m = -1e30f;
#pragma unroll
        for (int j = 0; j < 16; ++j) {
            sv[j] = wm.sc[i * 16 + j] + wm.P[i * 32 + (i - j + 15)];
            m = fmaxf(m, sv[j]);
        }
        float sum = 0.f;
#pragma unroll
        for (int j = 0; j < 16; ++j) { sv[j] = expf(sv[j] - m); sum += sv[j]; }
        float inv = 1.f / sum;
#pragma unroll
        for (int j = 0; j < 16; ++j) {
            wm.attn[i * 32 + j]      = f2bf(sv[j] * inv);
            wm.attn[i * 32 + 16 + j] = f2bf(0.f);
        }
    }

    // out = attn @ v  (K padded to 32 with zeros); 4 N-tiles of 16 over d
#pragma unroll
    for (int dt = 0; dt < 4; ++dt) {
        f32x8 accO = __builtin_amdgcn_wmma_f32_16x16x32_bf16(
            false, frag_a(&wm.attn[lc * 32], lh),
            false, frag_b(&wm.vT[(dt * 16 + lc) * 32], lh),
            (short)0, zero8(), false, false);
#pragma unroll
        for (int vi = 0; vi < 8; ++vi) {
            int f = vi + lh * 8;
            int d = dt * 16 + lc;
            out[(size_t)(f * NSPAT + s) * DIMC + head * HDIM + d] = f2bf(accO[vi]);
        }
    }
}

// ---------------------------------------------------------------------------
extern "C" void kernel_launch(void* const* d_in, const int* in_sizes, int n_in,
                              void* d_out, int out_size, void* d_ws, size_t ws_size,
                              hipStream_t stream) {
    (void)in_sizes; (void)n_in; (void)out_size; (void)ws_size;
    const float* x       = (const float*)d_in[0];
    const float* qkv_w   = (const float*)d_in[1];
    const float* qkv_b   = (const float*)d_in[2];
    const float* proj_w  = (const float*)d_in[3];
    const float* proj_b  = (const float*)d_in[4];
    const float* rel_pos = (const float*)d_in[5];
    const float* ln1_w   = (const float*)d_in[6];
    const float* ln1_b   = (const float*)d_in[7];
    const float* ln2_w   = (const float*)d_in[8];
    const float* ln2_b   = (const float*)d_in[9];
    const float* mlp_w1  = (const float*)d_in[10];
    const float* mlp_b1  = (const float*)d_in[11];
    const float* mlp_w2  = (const float*)d_in[12];
    const float* mlp_b2  = (const float*)d_in[13];
    float* out = (float*)d_out;

    char* ws = (char*)d_ws;
    size_t off = 0;
    bf16*  bigBuf  = (bf16*)(ws + off); off += (size_t)NTOK * MLPD * 2;   // qkv, then mlp hidden
    bf16*  actBuf  = (bf16*)(ws + off); off += (size_t)NTOK * DIMC * 2;   // xn -> attn_out -> ln2
    float* projBuf = (float*)(ws + off); off += (size_t)NTOK * DIMC * 4;
    bf16*  qkvw_h  = (bf16*)(ws + off); off += (size_t)3 * DIMC * DIMC * 2;
    bf16*  projw_h = (bf16*)(ws + off); off += (size_t)DIMC * DIMC * 2;
    bf16*  w1_h    = (bf16*)(ws + off); off += (size_t)MLPD * DIMC * 2;
    bf16*  w2_h    = (bf16*)(ws + off); off += (size_t)DIMC * MLPD * 2;

    // weights -> bf16
    cvt_kernel<<<(3 * DIMC * DIMC + 255) / 256, 256, 0, stream>>>(qkv_w, qkvw_h, 3 * DIMC * DIMC);
    cvt_kernel<<<(DIMC * DIMC + 255) / 256, 256, 0, stream>>>(proj_w, projw_h, DIMC * DIMC);
    cvt_kernel<<<(MLPD * DIMC + 255) / 256, 256, 0, stream>>>(mlp_w1, w1_h, MLPD * DIMC);
    cvt_kernel<<<(DIMC * MLPD + 255) / 256, 256, 0, stream>>>(mlp_w2, w2_h, DIMC * MLPD);

    // LN1 -> bf16
    ln_kernel<<<NTOK / 8, 256, 0, stream>>>(x, ln1_w, ln1_b, actBuf);
    // qkv = xn @ qkv_w^T + b   (bf16 out)
    gemm_bf16<0><<<dim3(3 * DIMC / 128, NTOK / 128), 256, 0, stream>>>(
        actBuf, qkvw_h, qkv_b, nullptr, bigBuf, 3 * DIMC, DIMC);
    // temporal attention -> actBuf (bf16)
    attn_kernel<<<(NSPAT * NHEAD) / 4, 128, 0, stream>>>(bigBuf, rel_pos, actBuf);
    // proj (f32 out for LN2)
    gemm_bf16<1><<<dim3(DIMC / 128, NTOK / 128), 256, 0, stream>>>(
        actBuf, projw_h, proj_b, nullptr, projBuf, DIMC, DIMC);
    // LN2 -> bf16
    ln_kernel<<<NTOK / 8, 256, 0, stream>>>(projBuf, ln2_w, ln2_b, actBuf);
    // mlp1 + GELU (bf16 hidden)
    gemm_bf16<2><<<dim3(MLPD / 128, NTOK / 128), 256, 0, stream>>>(
        actBuf, w1_h, mlp_b1, nullptr, bigBuf, MLPD, DIMC);
    // mlp2 + bias + residual(x) -> f32 out
    gemm_bf16<3><<<dim3(DIMC / 128, NTOK / 128), 256, 0, stream>>>(
        bigBuf, w2_h, mlp_b2, x, out, DIMC, MLPD);
}
